// SpectralConv1d_plus_64948495450322
// MI455X (gfx1250) — matrix operand order repaired and checked by reading, not compiled, not run on hardware
//
#include <hip/hip_runtime.h>
#include <hip/hip_bf16.h>

// ---------------------------------------------------------------------------
// SpectralConv1d (FNO layer) for MI455X / gfx1250.
//
//   Stage A: x_ft  = Xrows[1024 x 16384] @ Fwd[16384 x 1024]   (bf16 WMMA, f32 acc)
//   Stage B: per-mode complex 32x32 einsum + width-5 freq conv  (f32 VALU, <1% flops)
//   Stage C: y     = Coef[1024 x 1024]  @ Inv[1024 x 16384]    (bf16 WMMA, f32 acc)
//
// GEMMs: per-wave 16x64 tile (4 f32 v8f accumulators). The A row-tile is
// shared by all 8 waves of a block -> staged through LDS with
// global_load_async_to_lds_b128 (ASYNCcnt) + s_wait_asynccnt + barrier,
// double-buffered.  B fragments stream from the L2-resident bf16 basis.
//
// Workspace (~106 MB): Xpack 32M | Fwd 32M | x_ft 4M | out1 4M | Coef 2M | Inv 32M
// ---------------------------------------------------------------------------

typedef __attribute__((ext_vector_type(16))) __bf16 v16bf;
typedef __attribute__((ext_vector_type(8)))  float  v8f;

#define BSZ    32
#define NLEN   16384
#define CIN    32
#define COUT   32
#define NMODES 512
#define FMIX   5
#define KCONV  (NMODES - FMIX + 1)   /* 508 valid conv outputs */

__device__ __forceinline__ float two_pi_over_n() { return 6.283185307179586f / (float)NLEN; }

// --- WMMA fragment coordinate maps (ISA 7.12.2, 16-bit operands, wave32) ----
__device__ __forceinline__ void a_frag_coords(int f, int& M, int& K) {
  int lane = f >> 4, e = f & 15;
  if (lane < 16) { M = lane;      K = (e < 8) ? e     : e + 8;  }
  else           { M = lane - 16; K = (e < 8) ? e + 8 : e + 16; }
}
__device__ __forceinline__ void b_frag_coords(int f, int& K, int& N) {
  int lane = f >> 4, e = f & 15;
  if (lane < 16) { N = lane;      K = (e < 8) ? e     : e + 8;  }
  else           { N = lane - 16; K = (e < 8) ? e + 8 : e + 16; }
}

// --- async A-tile staging helpers ------------------------------------------
__device__ __forceinline__ void issue_async_tile(const char* gsrc, unsigned lds_off) {
  // one 1 KB tile, 64 lanes x 16 B:  LDS[lds_off] <= MEM[gsrc]
  asm volatile("global_load_async_to_lds_b128 %0, %1, off"
               :: "v"(lds_off), "v"(gsrc) : "memory");
}
__device__ __forceinline__ void wait_async_publish() {
  asm volatile("s_wait_asynccnt 0x0" ::: "memory");
  __syncthreads();
}

// ---------------------------------------------------------------------------
// 1) Repack x [B,N,Cin] -> A-fragments of rows (b,i), K = n.  64 rt x 512 kt tiles.
__global__ void pack_x_kernel(const float* __restrict__ x, __bf16* __restrict__ Xp) {
  int idx = blockIdx.x * blockDim.x + threadIdx.x;     // 64*512*512
  int f = idx & 511, tile = idx >> 9;
  int kt = tile & 511, rt = tile >> 9;
  int M, K; a_frag_coords(f, M, K);
  int row = rt * 16 + M;                               // row = b*32 + i
  int b = row >> 5, i = row & 31;
  int n = kt * 32 + K;
  Xp[idx] = (__bf16)x[(b * NLEN + n) * CIN + i];
}

// 2) Forward DFT basis, B-fragments: rows K=n, cols = 2k(+re/im). 512 kt x 64 ct.
__global__ void gen_fwd_basis_kernel(__bf16* __restrict__ Fb) {
  int idx = blockIdx.x * blockDim.x + threadIdx.x;     // 512*64*512
  int f = idx & 511, tile = idx >> 9;
  int ct = tile & 63, kt = tile >> 6;
  int K, N; b_frag_coords(f, K, N);
  int n = kt * 32 + K;
  int col = ct * 16 + N;
  int k = col >> 1;
  float th = (float)((n * k) & (NLEN - 1)) * two_pi_over_n();
  float v = (col & 1) ? -__sinf(th) : __cosf(th);
  Fb[idx] = (__bf16)v;
}

// 3) Stage-A GEMM: x_ft[1024 x 1024] = Xpack @ Fwd.
//    16x64 per wave; shared A tile double-buffered in LDS via async loads.
__global__ __launch_bounds__(256) void gemmA_kernel(const __bf16* __restrict__ Ax,
                                                    const __bf16* __restrict__ Bf,
                                                    float* __restrict__ xft) {
  const int NKT = 512, NCT = 64;
  __shared__ __align__(16) __bf16 atile[2 * 512];       // 2 x 1 KB
  int lane = threadIdx.x & 31, wave = threadIdx.x >> 5;
  int rt = blockIdx.x >> 1;                             // 0..63
  int ctBase = (blockIdx.x & 1) * 32 + wave * 4;        // 4 col tiles per wave
  const v16bf* Bp  = (const v16bf*)Bf;
  const v16bf* Ash = (const v16bf*)atile;
  const char*  Ab  = (const char*)Ax;
  unsigned lds0 = (unsigned)(uintptr_t)(void*)atile;    // LDS byte offset of atile
  v8f acc[4] = {};

  if (threadIdx.x < 64)
    issue_async_tile(Ab + (size_t)(rt * NKT) * 1024 + threadIdx.x * 16,
                     lds0 + threadIdx.x * 16);
  wait_async_publish();

  for (int kt = 0; kt < NKT; ++kt) {
    int cur = kt & 1;
    if (kt + 1 < NKT && threadIdx.x < 64)
      issue_async_tile(Ab + (size_t)(rt * NKT + kt + 1) * 1024 + threadIdx.x * 16,
                       lds0 + (unsigned)(((kt + 1) & 1) * 1024 + threadIdx.x * 16));
    v16bf a = Ash[cur * 32 + lane];
    if (kt + 1 < NKT)
      __builtin_prefetch(&Bp[((kt + 1) * NCT + ctBase) * 32 + lane], 0, 2);
#pragma unroll
    for (int c = 0; c < 4; ++c) {
      v16bf b = Bp[((size_t)kt * NCT + (ctBase + c)) * 32 + lane];
      acc[c] = __builtin_amdgcn_wmma_f32_16x16x32_bf16(false, a, false, b,
                                                       (short)0, acc[c], false, false);
    }
    wait_async_publish();
  }

  int mb = (lane >= 16) ? 8 : 0, nc = lane & 15;
#pragma unroll
  for (int c = 0; c < 4; ++c) {
    int col = (ctBase + c) * 16 + nc;
#pragma unroll
    for (int r = 0; r < 8; ++r)
      xft[(rt * 16 + mb + r) * 1024 + col] = acc[c][r];
  }
}

// 4) Per-mode complex einsum  out1[b,o,k] = sum_i x_ft[b,i,k] * W[i,o,k]
__global__ void mode_mix_kernel(const float* __restrict__ xft,
                                const float* __restrict__ wr,
                                const float* __restrict__ wi,
                                float* __restrict__ out1) {
  int idx = blockIdx.x * blockDim.x + threadIdx.x;      // B*Cout*512
  int k = idx & (NMODES - 1);
  int t = idx >> 9, o = t & 31, b = t >> 5;
  float ar = 0.f, ai = 0.f;
#pragma unroll 4
  for (int i = 0; i < CIN; ++i) {
    float xr = xft[(b * CIN + i) * 1024 + 2 * k];
    float xi = xft[(b * CIN + i) * 1024 + 2 * k + 1];
    float Wr = wr[(i * COUT + o) * NMODES + k];
    float Wi = wi[(i * COUT + o) * NMODES + k];
    ar += xr * Wr - xi * Wi;
    ai += xr * Wi + xi * Wr;
  }
  out1[idx * 2 + 0] = ar;
  out1[idx * 2 + 1] = ai;
}

// 5) Freq conv (width 5, mixes Cout) + irfft scaling, packed as A-fragments.
__global__ void conv_pack_kernel(const float* __restrict__ out1,
                                 const float* __restrict__ kr,
                                 const float* __restrict__ ki,
                                 __bf16* __restrict__ Coef) {
  int idx = blockIdx.x * blockDim.x + threadIdx.x;      // 64*32*512
  int f = idx & 511, tile = idx >> 9;
  int kt = tile & 31, rt = tile >> 5;
  int M, K; a_frag_coords(f, M, K);
  int row = rt * 16 + M;                                // (b,o)
  int b = row >> 5, o = row & 31;
  int col = kt * 32 + K;
  int k = col >> 1;
  bool isRe = !(col & 1);
  float v = 0.f;
  if (k < KCONV) {
    const float* kern = isRe ? kr : ki;
    int part = isRe ? 0 : 1;
    float s = 0.f;
#pragma unroll
    for (int w = 0; w < FMIX; ++w)
      for (int i = 0; i < COUT; ++i)
        s += out1[((b * COUT + i) * NMODES + (k + w)) * 2 + part] *
             kern[(w * COUT + i) * COUT + o];
    float alpha = (k == 0 ? 1.f : 2.f) / (float)NLEN;
    v = isRe ? alpha * s : -alpha * s;
  }
  Coef[idx] = (__bf16)v;
}

// 6) Inverse basis, B-fragments: rows K = 2k(+cos/sin), cols n. 32 kt x 1024 nt.
__global__ void gen_inv_basis_kernel(__bf16* __restrict__ Ib) {
  int idx = blockIdx.x * blockDim.x + threadIdx.x;      // 32*1024*512
  int f = idx & 511, tile = idx >> 9;
  int nt = tile & 1023, kt = tile >> 10;
  int K, N; b_frag_coords(f, K, N);
  int kk = kt * 32 + K;
  int k = kk >> 1;
  int n = nt * 16 + N;
  float th = (float)((n * k) & (NLEN - 1)) * two_pi_over_n();
  float v = (kk & 1) ? __sinf(th) : __cosf(th);
  Ib[idx] = (__bf16)v;
}

// 7) Stage-C GEMM: Y[1024 x 16384] = Coef @ Inv, stored transposed to [B,N,Cout].
__global__ __launch_bounds__(256) void gemmC_kernel(const __bf16* __restrict__ Ac,
                                                    const __bf16* __restrict__ Bi,
                                                    float* __restrict__ y) {
  const int NKT = 32, NCT = 1024;
  __shared__ __align__(16) __bf16 atile[2 * 512];
  int lane = threadIdx.x & 31, wave = threadIdx.x >> 5;
  int blk = blockIdx.x;                                 // 0..2047
  int rt  = blk >> 5;                                   // 0..63
  int ctBase = (blk & 31) * 32 + wave * 4;              // 4 col tiles per wave
  const v16bf* Bp  = (const v16bf*)Bi;
  const v16bf* Ash = (const v16bf*)atile;
  const char*  Ab  = (const char*)Ac;
  unsigned lds0 = (unsigned)(uintptr_t)(void*)atile;
  v8f acc[4] = {};

  if (threadIdx.x < 64)
    issue_async_tile(Ab + (size_t)(rt * NKT) * 1024 + threadIdx.x * 16,
                     lds0 + threadIdx.x * 16);
  wait_async_publish();

  for (int kt = 0; kt < NKT; ++kt) {
    int cur = kt & 1;
    if (kt + 1 < NKT && threadIdx.x < 64)
      issue_async_tile(Ab + (size_t)(rt * NKT + kt + 1) * 1024 + threadIdx.x * 16,
                       lds0 + (unsigned)(((kt + 1) & 1) * 1024 + threadIdx.x * 16));
    v16bf a = Ash[cur * 32 + lane];
    if (kt + 1 < NKT)
      __builtin_prefetch(&Bp[((kt + 1) * NCT + ctBase) * 32 + lane], 0, 2);
#pragma unroll
    for (int c = 0; c < 4; ++c) {
      v16bf b = Bp[((size_t)kt * NCT + (ctBase + c)) * 32 + lane];
      acc[c] = __builtin_amdgcn_wmma_f32_16x16x32_bf16(false, a, false, b,
                                                       (short)0, acc[c], false, false);
    }
    wait_async_publish();
  }

  int mb = (lane >= 16) ? 8 : 0, nc = lane & 15;
#pragma unroll
  for (int c = 0; c < 4; ++c) {
    int n = (ctBase + c) * 16 + nc;
#pragma unroll
    for (int r = 0; r < 8; ++r) {
      int row = rt * 16 + mb + r;                       // (b,o)
      int b = row >> 5, o = row & 31;
      y[(b * NLEN + n) * COUT + o] = acc[c][r];
    }
  }
}

// ---------------------------------------------------------------------------
extern "C" void kernel_launch(void* const* d_in, const int* in_sizes, int n_in,
                              void* d_out, int out_size, void* d_ws, size_t ws_size,
                              hipStream_t stream) {
  (void)in_sizes; (void)n_in; (void)out_size; (void)ws_size;
  const float* x      = (const float*)d_in[0];
  const float* w_real = (const float*)d_in[1];
  const float* w_imag = (const float*)d_in[2];
  const float* k_real = (const float*)d_in[3];
  const float* k_imag = (const float*)d_in[4];
  float* y = (float*)d_out;

  char* ws = (char*)d_ws;
  const size_t SZ_XP   = (size_t)1024 * NLEN * 2;             // 32 MB bf16
  const size_t SZ_FWD  = (size_t)NLEN * 1024 * 2;             // 32 MB bf16
  const size_t SZ_XFT  = (size_t)1024 * 1024 * 4;             //  4 MB f32
  const size_t SZ_OUT1 = (size_t)BSZ * COUT * NMODES * 2 * 4; //  4 MB f32
  const size_t SZ_COEF = (size_t)1024 * 1024 * 2;             //  2 MB bf16
  __bf16* Xp   = (__bf16*)(ws);
  __bf16* Fwd  = (__bf16*)(ws + SZ_XP);
  float*  xft  = (float*) (ws + SZ_XP + SZ_FWD);
  float*  out1 = (float*) (ws + SZ_XP + SZ_FWD + SZ_XFT);
  __bf16* Coef = (__bf16*)(ws + SZ_XP + SZ_FWD + SZ_XFT + SZ_OUT1);
  __bf16* Inv  = (__bf16*)(ws + SZ_XP + SZ_FWD + SZ_XFT + SZ_OUT1 + SZ_COEF);

  pack_x_kernel       <<<(64 * 512 * 512) / 256, 256, 0, stream>>>(x, Xp);
  gen_fwd_basis_kernel<<<(512 * 64 * 512) / 256, 256, 0, stream>>>(Fwd);
  gemmA_kernel        <<<128, 256, 0, stream>>>(Xp, Fwd, xft);
  mode_mix_kernel     <<<(BSZ * COUT * NMODES) / 256, 256, 0, stream>>>(xft, w_real, w_imag, out1);
  conv_pack_kernel    <<<(64 * 32 * 512) / 256, 256, 0, stream>>>(out1, k_real, k_imag, Coef);
  gen_inv_basis_kernel<<<(32 * 1024 * 512) / 256, 256, 0, stream>>>(Inv);
  gemmC_kernel        <<<2048, 256, 0, stream>>>(Coef, Inv, y);
}
